// Bottleneck_1w1a_49460843381708
// MI455X (gfx1250) — 8-bit matrix operand order repaired and checked by reading, compiled, not run on hardware
//
#include <hip/hip_runtime.h>
#include <cstdint>


typedef int   v16i __attribute__((ext_vector_type(16)));
typedef float v8f  __attribute__((ext_vector_type(8)));

static constexpr int NB   = 32;
static constexpr int HHc  = 28;
static constexpr int WWc  = 28;
static constexpr int HW   = 784;
static constexpr int CIN  = 1024;
static constexpr int P    = 256;
static constexpr int C3   = 1024;
static constexpr int MTOT = NB * HW;

static constexpr size_t OFF_ACT0 = 0;
static constexpr size_t OFF_ACT1 = OFF_ACT0 + (size_t)MTOT * CIN;
static constexpr size_t OFF_ACT2 = OFF_ACT1 + (size_t)MTOT * P;
static constexpr size_t OFF_BW1  = OFF_ACT2 + (size_t)MTOT * P;
static constexpr size_t OFF_BW2  = OFF_BW1 + (size_t)CIN * P;
static constexpr size_t OFF_BW3  = OFF_BW2 + (size_t)(9 * P) * P;
static constexpr size_t OFF_SC   = OFF_BW3 + (size_t)P * C3;

__device__ __forceinline__ int8_t fp8_sign(float v) {
    return v > 0.f ? (int8_t)0x38 : (v < 0.f ? (int8_t)0xB8 : (int8_t)0);
}

__global__ void __launch_bounds__(256)
prep_scales(const float* __restrict__ a1, const float* __restrict__ g1, const float* __restrict__ b1,
            const float* __restrict__ a2, const float* __restrict__ g2, const float* __restrict__ b2,
            const float* __restrict__ a3, const float* __restrict__ g3, const float* __restrict__ b3,
            float* __restrict__ sc)
{
    const int i = blockIdx.x * 256 + threadIdx.x;
    const float rs = rsqrtf(1.0f + 1e-5f);
    if (i < 256) {
        sc[i]        = a1[i] * g1[i] * rs;
        sc[256 + i]  = b1[i];
        sc[512 + i]  = a2[i] * g2[i] * rs;
        sc[768 + i]  = b2[i];
    }
    if (i < 1024) {
        sc[1024 + i] = a3[i] * g3[i] * rs;
        sc[2048 + i] = b3[i];
    }
}

__global__ void __launch_bounds__(256)
binarize_weights(const float* __restrict__ w, int8_t* __restrict__ bfrag,
                 int K, int nTiles, int mode)
{
    __shared__ float red[256];
    const int o = blockIdx.x;
    const int t = threadIdx.x;
    const float* wf = w + (size_t)o * K;

    float s = 0.f;
    for (int k = t; k < K; k += 256) s += wf[k];
    red[t] = s;
    __syncthreads();
    for (int st = 128; st > 0; st >>= 1) {
        if (t < st) red[t] += red[t + st];
        __syncthreads();
    }
    const float mean = red[0] / (float)K;

    for (int k = t; k < K; k += 256) {
        const int src = (mode == 1) ? ((k & 255) * 9 + (k >> 8)) : k;
        const int8_t sg = fp8_sign(wf[src] - mean);
        const int kt = k >> 7, kin = k & 127;
        const int blk32  = kin >> 5;
        const int within = kin & 31;
        const int khalf  = within >> 4;
        const int rem    = within & 15;
        const int vv     = (blk32 << 2) | (rem >> 2);
        const int byte   = rem & 3;
        const int lane   = (khalf << 4) | (o & 15);
        bfrag[(size_t)(kt * nTiles + (o >> 4)) * 2048 +
              (size_t)lane * 64 + vv * 4 + byte] = sg;
    }
}

__global__ void __launch_bounds__(256)
binarize_act(const float* __restrict__ x, int8_t* __restrict__ act0)
{
    const size_t idx = (size_t)blockIdx.x * 256 + threadIdx.x;
    if (idx >= (size_t)NB * CIN * HW) return;
    const int n   = (int)(idx / ((size_t)CIN * HW));
    const int rem = (int)(idx - (size_t)n * CIN * HW);
    const int c   = rem / HW;
    const int hw  = rem - c * HW;
    act0[(size_t)(n * HW + hw) * CIN + c] = fp8_sign(x[idx]);
}

__device__ __forceinline__ int a_off128(int v, int half) {
    return ((v >> 2) << 5) + (half << 4) + ((v & 3) << 2);
}

__global__ void __launch_bounds__(256)
gemm_fp8_sign(const int8_t* __restrict__ act, const int8_t* __restrict__ bfrag,
              int K, int Nout,
              const float* __restrict__ sA, const float* __restrict__ sB,
              int8_t* __restrict__ outAct)
{
    const int lane = threadIdx.x & 31;
    const int wid  = (blockIdx.x * 256 + threadIdx.x) >> 5;
    const int nPairs = Nout >> 5;
    const int mTile  = wid / nPairs;
    const int nPair  = wid - mTile * nPairs;
    if (mTile >= (MTOT >> 4)) return;

    v8f acc0 = {}, acc1 = {};
    const int half = lane >> 4;
    const int mrow = (mTile << 4) + (lane & 15);
    const int8_t* __restrict__ arow = act + (size_t)mrow * K;
    const int nTiles = Nout >> 4;
    const int ktn = K >> 7;
    for (int kt = 0; kt < ktn; ++kt) {
        v16i a;
#pragma unroll
        for (int v = 0; v < 16; ++v)
            a[v] = *(const int*)(arow + (kt << 7) + a_off128(v, half));
        const v16i b0 = ((const v16i*)(bfrag + (size_t)(kt * nTiles + (nPair << 1)) * 2048))[lane];
        const v16i b1 = ((const v16i*)(bfrag + (size_t)(kt * nTiles + (nPair << 1) + 1) * 2048))[lane];
        __builtin_prefetch(arow + ((kt + 1) << 7), 0, 1);
        acc0 = __builtin_amdgcn_wmma_f32_16x16x128_fp8_fp8(a, b0, (short)0, acc0, false, false);
        acc1 = __builtin_amdgcn_wmma_f32_16x16x128_fp8_fp8(a, b1, (short)0, acc1, false, false);
    }

    const int n0 = (nPair << 5) + (lane & 15);
    const float sa0 = sA[n0],      sb0 = sB[n0];
    const float sa1 = sA[n0 + 16], sb1 = sB[n0 + 16];
#pragma unroll
    for (int r = 0; r < 8; ++r) {
        const int m = (mTile << 4) + (half << 3) + r;
        outAct[(size_t)m * Nout + n0]      = fp8_sign(acc0[r] * sa0 + sb0);
        outAct[(size_t)m * Nout + n0 + 16] = fp8_sign(acc1[r] * sa1 + sb1);
    }
}

__global__ void __launch_bounds__(256)
conv3x3_fp8_sign(const int8_t* __restrict__ act,
                 const int8_t* __restrict__ bfrag,
                 const float* __restrict__ sA, const float* __restrict__ sB,
                 int8_t* __restrict__ outAct)
{
    const int lane = threadIdx.x & 31;
    const int wid  = (blockIdx.x * 256 + threadIdx.x) >> 5;
    const int nPairs = P >> 5;
    const int mTile  = wid / nPairs;
    const int nPair  = wid - mTile * nPairs;
    if (mTile >= (MTOT >> 4)) return;

    v8f acc0 = {}, acc1 = {};
    const int half = lane >> 4;
    const int mrow = (mTile << 4) + (lane & 15);
    const int n_img = mrow / HW;
    const int hw    = mrow - n_img * HW;
    const int h     = hw / WWc;
    const int w     = hw - h * WWc;
    const int nTiles = P >> 4;

    for (int kt = 0; kt < 18; ++kt) {
        const int tap = kt >> 1;
        const int hh  = h + tap / 3 - 1;
        const int ww  = w + tap % 3 - 1;
        const int cbase = (kt & 1) << 7;
        const bool valid = ((unsigned)hh < (unsigned)HHc) && ((unsigned)ww < (unsigned)WWc);
        const int8_t* __restrict__ base =
            act + ((size_t)(n_img * HW + hh * WWc + ww) * P + cbase);
        v16i a;
#pragma unroll
        for (int v = 0; v < 16; ++v)
            a[v] = valid ? *(const int*)(base + a_off128(v, half)) : 0;
        const v16i b0 = ((const v16i*)(bfrag + (size_t)(kt * nTiles + (nPair << 1)) * 2048))[lane];
        const v16i b1 = ((const v16i*)(bfrag + (size_t)(kt * nTiles + (nPair << 1) + 1) * 2048))[lane];
        acc0 = __builtin_amdgcn_wmma_f32_16x16x128_fp8_fp8(a, b0, (short)0, acc0, false, false);
        acc1 = __builtin_amdgcn_wmma_f32_16x16x128_fp8_fp8(a, b1, (short)0, acc1, false, false);
    }

    const int n0 = (nPair << 5) + (lane & 15);
    const float sa0 = sA[n0],      sb0 = sB[n0];
    const float sa1 = sA[n0 + 16], sb1 = sB[n0 + 16];
#pragma unroll
    for (int r = 0; r < 8; ++r) {
        const int m = (mTile << 4) + (half << 3) + r;
        outAct[(size_t)m * P + n0]      = fp8_sign(acc0[r] * sa0 + sb0);
        outAct[(size_t)m * P + n0 + 16] = fp8_sign(acc1[r] * sa1 + sb1);
    }
}

__global__ void __launch_bounds__(256)
gemm_fp8_final(const int8_t* __restrict__ act,
               const int8_t* __restrict__ bfrag,
               const float* __restrict__ sA, const float* __restrict__ sB,
               const float* __restrict__ x,
               float* __restrict__ out)
{
    const int lane = threadIdx.x & 31;
    const int wid  = (blockIdx.x * 256 + threadIdx.x) >> 5;
    const int nPairs = C3 >> 5;
    const int mTile  = wid / nPairs;
    const int nPair  = wid - mTile * nPairs;
    if (mTile >= (MTOT >> 4)) return;

    v8f acc0 = {}, acc1 = {};
    const int half = lane >> 4;
    const int mrow = (mTile << 4) + (lane & 15);
    const int8_t* __restrict__ arow = act + (size_t)mrow * P;
    const int nTiles = C3 >> 4;
#pragma unroll
    for (int kt = 0; kt < (P >> 7); ++kt) {
        v16i a;
#pragma unroll
        for (int v = 0; v < 16; ++v)
            a[v] = *(const int*)(arow + (kt << 7) + a_off128(v, half));
        const v16i b0 = ((const v16i*)(bfrag + (size_t)(kt * nTiles + (nPair << 1)) * 2048))[lane];
        const v16i b1 = ((const v16i*)(bfrag + (size_t)(kt * nTiles + (nPair << 1) + 1) * 2048))[lane];
        acc0 = __builtin_amdgcn_wmma_f32_16x16x128_fp8_fp8(a, b0, (short)0, acc0, false, false);
        acc1 = __builtin_amdgcn_wmma_f32_16x16x128_fp8_fp8(a, b1, (short)0, acc1, false, false);
    }

    const int c0 = (nPair << 5) + (lane & 15);
    const float sa0 = sA[c0],      sb0 = sB[c0];
    const float sa1 = sA[c0 + 16], sb1 = sB[c0 + 16];
#pragma unroll
    for (int r = 0; r < 8; ++r) {
        const int m = (mTile << 4) + (half << 3) + r;
        const int n_img = m / HW;
        const int hw    = m - n_img * HW;
        const size_t base = (size_t)n_img * ((size_t)C3 * HW) + hw;
        const size_t i0 = base + (size_t)c0 * HW;
        const size_t i1 = base + (size_t)(c0 + 16) * HW;
        float y0 = acc0[r] * sa0 + sb0 + x[i0];
        float y1 = acc1[r] * sa1 + sb1 + x[i1];
        out[i0] = fminf(1.f, fmaxf(-1.f, y0));
        out[i1] = fminf(1.f, fmaxf(-1.f, y1));
    }
}

extern "C" void kernel_launch(void* const* d_in, const int* in_sizes, int n_in,
                              void* d_out, int out_size, void* d_ws, size_t ws_size,
                              hipStream_t stream)
{
    const float* x  = (const float*)d_in[0];
    const float* w1 = (const float*)d_in[1];
    const float* a1 = (const float*)d_in[2];
    const float* g1 = (const float*)d_in[3];
    const float* b1 = (const float*)d_in[4];
    const float* w2 = (const float*)d_in[5];
    const float* a2 = (const float*)d_in[6];
    const float* g2 = (const float*)d_in[7];
    const float* b2 = (const float*)d_in[8];
    const float* w3 = (const float*)d_in[9];
    const float* a3 = (const float*)d_in[10];
    const float* g3 = (const float*)d_in[11];
    const float* b3 = (const float*)d_in[12];

    uint8_t* ws = (uint8_t*)d_ws;
    int8_t* act0 = (int8_t*)(ws + OFF_ACT0);
    int8_t* act1 = (int8_t*)(ws + OFF_ACT1);
    int8_t* act2 = (int8_t*)(ws + OFF_ACT2);
    int8_t* bw1  = (int8_t*)(ws + OFF_BW1);
    int8_t* bw2  = (int8_t*)(ws + OFF_BW2);
    int8_t* bw3  = (int8_t*)(ws + OFF_BW3);
    float*  sc   = (float*)(ws + OFF_SC);

    prep_scales<<<4, 256, 0, stream>>>(a1, g1, b1, a2, g2, b2, a3, g3, b3, sc);
    binarize_weights<<<256,  256, 0, stream>>>(w1, bw1, CIN,   P  >> 4, 0);
    binarize_weights<<<256,  256, 0, stream>>>(w2, bw2, 9 * P, P  >> 4, 1);
    binarize_weights<<<1024, 256, 0, stream>>>(w3, bw3, P,     C3 >> 4, 0);

    binarize_act<<<(int)(((size_t)NB * CIN * HW + 255) / 256), 256, 0, stream>>>(x, act0);

    gemm_fp8_sign<<<(MTOT >> 4) * (P >> 5) / 8, 256, 0, stream>>>(
        act0, bw1, CIN, P, sc, sc + 256, act1);

    conv3x3_fp8_sign<<<(MTOT >> 4) * (P >> 5) / 8, 256, 0, stream>>>(
        act1, bw2, sc + 512, sc + 768, act2);

    gemm_fp8_final<<<(MTOT >> 4) * (C3 >> 5) / 8, 256, 0, stream>>>(
        act2, bw3, sc + 1024, sc + 2048, x, (float*)d_out);
}
